// KAN_34754875359700
// MI455X (gfx1250) — matrix operand order repaired and checked by reading, not compiled, not run on hardware
//
#include <hip/hip_runtime.h>

#define SEQ 2048
#define DIM 1024
#define NH 4
#define HLEN 256
#define FFDIM 400
#define FFP 416          // FF padded to multiple of 32 (K granularity of the GEMM)
#define NCLS 19
#define EPSN 1e-3f

typedef unsigned short u16;
typedef __bf16 bf16_t;
typedef bf16_t v16bf __attribute__((ext_vector_type(16)));
typedef float v8f __attribute__((ext_vector_type(8)));
typedef unsigned int u32x4 __attribute__((ext_vector_type(4)));
typedef int v4i_t __attribute__((ext_vector_type(4)));

#if __has_builtin(__builtin_amdgcn_global_load_async_to_lds_b128)
#define USE_ASYNC_LDS 1
typedef __attribute__((address_space(1))) v4i_t as1_v4i;
typedef __attribute__((address_space(3))) v4i_t as3_v4i;
#define GASYNC_B128(g, l) __builtin_amdgcn_global_load_async_to_lds_b128( \
    (as1_v4i*)(g), (as3_v4i*)(l), 0, 0)
#if __has_builtin(__builtin_amdgcn_s_wait_asynccnt)
#define WAIT_ASYNC() __builtin_amdgcn_s_wait_asynccnt(0)
#else
#define WAIT_ASYNC() asm volatile("s_wait_asynccnt 0x0" ::: "memory")
#endif
#else
#define USE_ASYNC_LDS 0
#endif

__device__ __forceinline__ u16 f2bf(float f) {
    unsigned int u = __float_as_uint(f);
    unsigned int r = u + 0x7fffu + ((u >> 16) & 1u);   // round-to-nearest-even
    return (u16)(r >> 16);
}

__device__ __forceinline__ float block_reduce_sum(float v, float* sh) {
    int t = threadIdx.x;
    sh[t] = v; __syncthreads();
    for (int s = 128; s > 0; s >>= 1) { if (t < s) sh[t] += sh[t + s]; __syncthreads(); }
    float r = sh[0]; __syncthreads();
    return r;
}

__device__ __forceinline__ float block_reduce_max(float v, float* sh) {
    int t = threadIdx.x;
    sh[t] = v; __syncthreads();
    for (int s = 128; s > 0; s >>= 1) { if (t < s) sh[t] = fmaxf(sh[t], sh[t + s]); __syncthreads(); }
    float r = sh[0]; __syncthreads();
    return r;
}

// ---------------------------------------------------------------------------
// WMMA GEMM: C(MxN, f32 accum) = A(MxK bf16 row-major) * B, where B is passed
// TRANSPOSED: Bt is N x K row-major (Bt[n][k] = B[k][n]).  Double-buffered LDS
// filled with GLOBAL_LOAD_ASYNC_TO_LDS_B128 (ASYNCcnt) so the next tile's DMA
// overlaps the current tile's WMMAs; per-lane fragment reads are contiguous
// 16B/32B runs -> ds_load_b128.
// Optional per-column f32 bias, optional ReLU; stores f32 (Cf) and/or bf16 (Cb).
// Requires: M % 128 == 0, K % 32 == 0. N may be ragged (bounds-checked).
// ---------------------------------------------------------------------------
#define BM 128
#define BN 128
#define BK 32
#define TS_LD (BK + 8)    // 40 u16 -> 80B row stride (16B aligned)

union BFrag { v16bf v; u16 u[16]; };

__global__ __launch_bounds__(256)
void gemm_bf16_wmma(const u16* __restrict__ A, int lda,
                    const u16* __restrict__ Bt, int ldb,
                    float* __restrict__ Cf, u16* __restrict__ Cb, int ldc,
                    int M, int N, int K,
                    const float* __restrict__ bias, int relu)
{
    __shared__ u16 As[2][BM * TS_LD];
    __shared__ u16 Bs[2][BN * TS_LD];

    const int tid  = threadIdx.x;
    const int lane = tid & 31;
    const int wave = tid >> 5;
    const int wm   = wave & 3;     // 4 waves along M (4*32 = 128)
    const int wn   = wave >> 2;    // 2 waves along N (2*64 = 128)
    const int hf   = lane >> 4;    // half-wave select
    const int l16  = lane & 15;

    const int bm = blockIdx.y * BM;
    const int bn = blockIdx.x * BN;

    // staging coordinates: chunk cc = tid + c*256 (c = 0,1); row = tid>>2 + c*64,
    // kc = (tid&3)*8 is invariant.
    const int srow = tid >> 2;          // 0..63
    const int skc  = (tid & 3) * 8;     // 0,8,16,24
    const u16* aBase = A + (size_t)(bm + srow) * lda + skc;
    int brow0 = bn + srow;      if (brow0 >= N) brow0 = N - 1;   // clamp: junk only
    int brow1 = bn + srow + 64; if (brow1 >= N) brow1 = N - 1;   // feeds unstored cols
    const u16* bBase0 = Bt + (size_t)brow0 * ldb + skc;
    const u16* bBase1 = Bt + (size_t)brow1 * ldb + skc;
    const int ldsOff  = srow * TS_LD + skc;

    v8f acc[2][4];
    #pragma unroll
    for (int i = 0; i < 2; i++)
        #pragma unroll
        for (int j = 0; j < 4; j++)
            acc[i][j] = (v8f){0.f,0.f,0.f,0.f,0.f,0.f,0.f,0.f};

#if USE_ASYNC_LDS
    auto stage = [&](int k0, int buf) {
        GASYNC_B128(aBase + k0,                      &As[buf][ldsOff]);
        GASYNC_B128(aBase + (size_t)64 * lda + k0,   &As[buf][ldsOff + 64 * TS_LD]);
        GASYNC_B128(bBase0 + k0,                     &Bs[buf][ldsOff]);
        GASYNC_B128(bBase1 + k0,                     &Bs[buf][ldsOff + 64 * TS_LD]);
    };
    stage(0, 0);
#else
    u32x4 arg[2], brg[2];
    auto load_regs = [&](int k0) {
        arg[0] = *(const u32x4*)(aBase + k0);
        arg[1] = *(const u32x4*)(aBase + (size_t)64 * lda + k0);
        brg[0] = *(const u32x4*)(bBase0 + k0);
        brg[1] = *(const u32x4*)(bBase1 + k0);
    };
    auto store_regs = [&](int buf) {
        *(u32x4*)(&As[buf][ldsOff])               = arg[0];
        *(u32x4*)(&As[buf][ldsOff + 64 * TS_LD])  = arg[1];
        *(u32x4*)(&Bs[buf][ldsOff])               = brg[0];
        *(u32x4*)(&Bs[buf][ldsOff + 64 * TS_LD])  = brg[1];
    };
    load_regs(0);
    store_regs(0);
    __syncthreads();
#endif

    int buf = 0;
    for (int k0 = 0; k0 < K; k0 += BK) {
        const bool more = (k0 + BK) < K;
#if USE_ASYNC_LDS
        WAIT_ASYNC();          // this wave's async fills of As[buf]/Bs[buf] done
        __syncthreads();       // everyone's fills done; prior compute on buf^1 done
        if (more) stage(k0 + BK, buf ^ 1);
#else
        if (more) load_regs(k0 + BK);   // batched global loads overlap compute below
#endif
        const u16* as = As[buf];
        const u16* bs = Bs[buf];

        // A frag (16x32): lanes 0-15 hold K 0-7 & 16-23, lanes 16-31 K 8-15 & 24-31
        BFrag afr[2];
        #pragma unroll
        for (int i = 0; i < 2; i++) {
            const u16* ap = &as[(wm * 32 + i * 16 + l16) * TS_LD];
            #pragma unroll
            for (int t = 0; t < 8; t++) {
                afr[i].u[t]     = ap[hf * 8 + t];
                afr[i].u[8 + t] = ap[16 + hf * 8 + t];
            }
        }
        // B frag (32x16): lanes 0-15 hold K 0-7 & 16-23, lanes 16-31 K 8-15 & 24-31
        BFrag bfr[4];
        #pragma unroll
        for (int j = 0; j < 4; j++) {
            const u16* bp = &bs[(wn * 64 + j * 16 + l16) * TS_LD + hf * 8];
            #pragma unroll
            for (int t = 0; t < 16; t++) bfr[j].u[t] = bp[t + (t >> 3) * 8];
        }
        #pragma unroll
        for (int i = 0; i < 2; i++)
            #pragma unroll
            for (int j = 0; j < 4; j++)
                acc[i][j] = __builtin_amdgcn_wmma_f32_16x16x32_bf16(
                    false, afr[i].v, false, bfr[j].v, (short)0, acc[i][j], false, false);

#if !USE_ASYNC_LDS
        __syncthreads();
        if (more) { store_regs(buf ^ 1); __syncthreads(); }
#endif
        buf ^= 1;
    }

    // ---- epilogue: C 16x16 f32 layout (VGPR r -> row r + hf*8, col l16)
    #pragma unroll
    for (int i = 0; i < 2; i++) {
        #pragma unroll
        for (int j = 0; j < 4; j++) {
            int n = bn + wn * 64 + j * 16 + l16;
            if (n >= N) continue;
            float bv = bias ? bias[n] : 0.f;
            #pragma unroll
            for (int r = 0; r < 8; r++) {
                int m = bm + wm * 32 + i * 16 + hf * 8 + r;
                float v = acc[i][j][r] + bv;
                if (relu) v = fmaxf(v, 0.f);
                if (Cf) Cf[(size_t)m * ldc + n] = v;
                if (Cb) Cb[(size_t)m * ldc + n] = f2bf(v);
            }
        }
    }
}

// ---------------------------------------------------------------------------
// Support kernels
// ---------------------------------------------------------------------------
__global__ void embed_pos_k(const int* __restrict__ cw, const int* __restrict__ pos,
                            const float* __restrict__ we, float* __restrict__ Xf)
{
    int s = blockIdx.x;
    int wrow = cw[s];
    float p = (float)pos[s];
    for (int d = threadIdx.x; d < DIM; d += blockDim.x) {
        float angle = p * __powf(10000.f, -(float)d / (float)DIM);
        float pe = ((d & 1) == 0) ? __sinf(angle) : __cosf(angle);
        Xf[(size_t)s * DIM + d] = we[(size_t)wrow * DIM + d] + pe;
    }
}

__global__ void f32_to_bf16_k(const float* __restrict__ in, u16* __restrict__ out, int n)
{
    int i = blockIdx.x * blockDim.x + threadIdx.x;
    if (i < n) out[i] = f2bf(in[i]);
}

// N x K converts for the projections: rows c = h*HL+d over D, k over D.
// wk, wv are already (c, k) row-major -> straight convert; wq rows have stride 2D.
__global__ void prep_qkv_k(const float* __restrict__ wq, const float* __restrict__ wk,
                           const float* __restrict__ wv,
                           u16* __restrict__ Wq, u16* __restrict__ Wk, u16* __restrict__ Wv)
{
    int idx = blockIdx.x * blockDim.x + threadIdx.x;
    if (idx >= DIM * DIM) return;
    int c = idx / DIM;
    int k = idx % DIM;
    Wq[idx] = f2bf(wq[(size_t)c * (2 * DIM) + k]);
    Wk[idx] = f2bf(wk[idx]);
    Wv[idx] = f2bf(wv[idx]);
}

// qbias[c] = sum_k wq[c][D+k] * e[k]   (rank-1 term from concat(out, e⊗1))
__global__ void prep_qbias_k(const float* __restrict__ wq, const float* __restrict__ e,
                             float* __restrict__ qbias)
{
    __shared__ float sred[256];
    int c = blockIdx.x;
    const float* row = wq + (size_t)c * (2 * DIM) + DIM;
    float s = 0.f;
    for (int k = threadIdx.x; k < DIM; k += 256) s += row[k] * e[k];
    s = block_reduce_sum(s, sred);
    if (threadIdx.x == 0) qbias[c] = s;
}

// Wo is used as logical B (K x N) -> Bt[n][k] = Wo[k][n] (transpose once)
__global__ void prep_wo_k(const float* __restrict__ Wo, u16* __restrict__ Wot)
{
    int idx = blockIdx.x * blockDim.x + threadIdx.x;
    if (idx >= DIM * DIM) return;
    int n = idx / DIM, k = idx % DIM;
    Wot[idx] = f2bf(Wo[(size_t)k * DIM + n]);
}

// W1 (FF x D) -> Bt rows f (padded to FFP with zeros); b1 padded too
__global__ void prep_w1_k(const float* __restrict__ W1, const float* __restrict__ b1,
                          u16* __restrict__ W1n, float* __restrict__ b1p)
{
    int idx = blockIdx.x * blockDim.x + threadIdx.x;
    if (idx >= FFP * DIM) return;
    int f = idx / DIM, k = idx % DIM;
    W1n[idx] = (f < FFDIM) ? f2bf(W1[(size_t)f * DIM + k]) : (u16)0;
    if (k == 0) b1p[f] = (f < FFDIM) ? b1[f] : 0.f;
}

// W2 (D x FF) -> Bt (D x FFP): [n][f] = W2[n*FF+f], zero-pad f >= FF
__global__ void prep_w2_k(const float* __restrict__ W2, u16* __restrict__ W2n)
{
    int idx = blockIdx.x * blockDim.x + threadIdx.x;
    if (idx >= DIM * FFP) return;
    int n = idx / FFP, f = idx % FFP;
    W2n[idx] = (f < FFDIM) ? f2bf(W2[(size_t)n * FFDIM + f]) : (u16)0;
}

// bf16 transpose (rows x cols) -> (cols x rows)
__global__ __launch_bounds__(256)
void transpose_bf16_k(const u16* __restrict__ in, u16* __restrict__ out, int rows, int cols)
{
    __shared__ u16 tile[32][33];
    int bx = blockIdx.x * 32;   // col base
    int by = blockIdx.y * 32;   // row base
    int tx = threadIdx.x & 31, tg = threadIdx.x >> 5;
    for (int r = tg; r < 32; r += 8)
        tile[r][tx] = in[(size_t)(by + r) * cols + bx + tx];
    __syncthreads();
    for (int r = tg; r < 32; r += 8)
        out[(size_t)(bx + r) * rows + by + tx] = tile[tx][r];
}

// row softmax of scale*scores - 1e5*I, write bf16 probs
__global__ void softmax_row_k(const float* __restrict__ sc, u16* __restrict__ pr, float scale)
{
    __shared__ float sred[256];
    int row = blockIdx.x;
    const float* src = sc + (size_t)row * SEQ;
    u16* dst = pr + (size_t)row * SEQ;
    float mx = -1e30f;
    for (int c = threadIdx.x; c < SEQ; c += 256) {
        float v = src[c] * scale - (c == row ? 1e5f : 0.f);
        mx = fmaxf(mx, v);
    }
    mx = block_reduce_max(mx, sred);
    float sum = 0.f;
    for (int c = threadIdx.x; c < SEQ; c += 256) {
        float v = src[c] * scale - (c == row ? 1e5f : 0.f);
        sum += __expf(v - mx);
    }
    sum = block_reduce_sum(sum, sred);
    float inv = 1.f / sum;
    for (int c = threadIdx.x; c < SEQ; c += 256) {
        float v = src[c] * scale - (c == row ? 1e5f : 0.f);
        dst[c] = f2bf(__expf(v - mx) * inv);
    }
}

// add + layernorm with unbiased std, eps added to sigma; write f32 + bf16
__global__ void add_norm_k(const float* __restrict__ z1, const float* __restrict__ z2,
                           const float* __restrict__ a, const float* __restrict__ b,
                           float* __restrict__ outf, u16* __restrict__ outb)
{
    __shared__ float sred[256];
    int row = blockIdx.x;
    const float* p1 = z1 + (size_t)row * DIM;
    const float* p2 = z2 + (size_t)row * DIM;
    float s1 = 0.f, s2 = 0.f;
    for (int d = threadIdx.x; d < DIM; d += 256) {
        float z = p1[d] + p2[d];
        s1 += z; s2 += z * z;
    }
    s1 = block_reduce_sum(s1, sred);
    s2 = block_reduce_sum(s2, sred);
    float mu = s1 / DIM;
    float var = (s2 - s1 * s1 / DIM) / (DIM - 1);
    float inv = 1.f / (sqrtf(fmaxf(var, 0.f)) + EPSN);
    for (int d = threadIdx.x; d < DIM; d += 256) {
        float z = p1[d] + p2[d];
        float v = (z - mu) * inv * a[d] + b[d];
        outf[(size_t)row * DIM + d] = v;
        outb[(size_t)row * DIM + d] = f2bf(v);
    }
}

// attention pooling: per column d, w = softmax_s(tanh(A[s,d])) (tanh bounded -> no max pass)
__global__ void pool_col_k(const float* __restrict__ Afp, const float* __restrict__ X,
                           float* __restrict__ pooled)
{
    __shared__ float sred[256];
    int d = blockIdx.x;
    float se = 0.f, sx = 0.f;
    for (int s = threadIdx.x; s < SEQ; s += 256) {
        float e = __expf(tanhf(Afp[(size_t)s * DIM + d]));
        se += e;
        sx += X[(size_t)s * DIM + d] * e;
    }
    se = block_reduce_sum(se, sred);
    sx = block_reduce_sum(sx, sred);
    if (threadIdx.x == 0) pooled[d] = sx / se;
}

__global__ void final_fc_k(const float* __restrict__ W, const float* __restrict__ bias,
                           const float* __restrict__ p0, const float* __restrict__ p1,
                           const float* __restrict__ rel, float* __restrict__ out)
{
    __shared__ float sred[256];
    int c = blockIdx.x;
    const float* w = W + (size_t)c * 3 * DIM;
    float s = 0.f;
    for (int j = threadIdx.x; j < 3 * DIM; j += 256) {
        float v = (j < DIM) ? p0[j] : ((j < 2 * DIM) ? p1[j - DIM] : rel[j - 2 * DIM]);
        s += w[j] * v;
    }
    s = block_reduce_sum(s, sred);
    if (threadIdx.x == 0) out[c] = s + bias[c];
}

// ---------------------------------------------------------------------------
extern "C" void kernel_launch(void* const* d_in, const int* in_sizes, int n_in,
                              void* d_out, int out_size, void* d_ws, size_t ws_size,
                              hipStream_t stream)
{
    const int*   contxtWords = (const int*)d_in[0];
    const float* e1          = (const float*)d_in[1];
    const float* e2          = (const float*)d_in[2];
    const int*   e1p         = (const int*)d_in[3];
    const int*   e2p         = (const int*)d_in[4];
    const float* relation    = (const float*)d_in[5];
    const float* word_embed  = (const float*)d_in[6];
    // l0/l1 dict leaves in jax pytree (sorted-key) order: W1,W2,Wo,a2,b1,b2,c2,wk,wq,wv
    struct Lay { const float *W1,*W2,*Wo,*a2,*b1,*b2,*c2,*wk,*wq,*wv; } L[2];
    for (int s = 0; s < 2; s++) {
        int b = 7 + s * 10;
        L[s].W1 = (const float*)d_in[b + 0];
        L[s].W2 = (const float*)d_in[b + 1];
        L[s].Wo = (const float*)d_in[b + 2];
        L[s].a2 = (const float*)d_in[b + 3];
        L[s].b1 = (const float*)d_in[b + 4];
        L[s].b2 = (const float*)d_in[b + 5];
        L[s].c2 = (const float*)d_in[b + 6];
        L[s].wk = (const float*)d_in[b + 7];
        L[s].wq = (const float*)d_in[b + 8];
        L[s].wv = (const float*)d_in[b + 9];
    }
    const float* att_W0 = (const float*)d_in[27];
    const float* att_b0 = (const float*)d_in[28];
    const float* sm_W   = (const float*)d_in[29];
    const float* sm_b   = (const float*)d_in[30];
    float* out = (float*)d_out;

    // ---- workspace carve-up
    char* wsb = (char*)d_ws;
    size_t off = 0;
    auto alloc = [&](size_t bytes) -> void* {
        void* p = wsb + off;
        off += (bytes + 255) & ~(size_t)255;
        return p;
    };
    float* Xf[2];
    Xf[0] = (float*)alloc((size_t)SEQ * DIM * 4);
    Xf[1] = (float*)alloc((size_t)SEQ * DIM * 4);
    u16*   Xb    = (u16*)  alloc((size_t)SEQ * DIM * 2);
    float* Yf    = (float*)alloc((size_t)SEQ * DIM * 4);
    u16*   Yb    = (u16*)  alloc((size_t)SEQ * DIM * 2);
    float* attnF = (float*)alloc((size_t)SEQ * DIM * 4);   // also reused for pool pre-activation
    float* ffnF  = (float*)alloc((size_t)SEQ * DIM * 4);
    u16*   Vq    = (u16*)  alloc((size_t)SEQ * DIM * 2);
    u16*   Vk    = (u16*)  alloc((size_t)SEQ * DIM * 2);
    u16*   Vp    = (u16*)  alloc((size_t)SEQ * DIM * 2);
    u16*   VpT   = (u16*)  alloc((size_t)DIM * SEQ * 2);
    u16*   headsB= (u16*)  alloc((size_t)SEQ * DIM * 2);
    u16*   Hb    = (u16*)  alloc((size_t)SEQ * FFP * 2);
    float* scoresF = (float*)alloc((size_t)SEQ * SEQ * 4); // one head at a time
    u16*   probsB  = (u16*)  alloc((size_t)SEQ * SEQ * 2);
    u16 *Wqn[2], *Wkn[2], *Wvn[2], *WoT[2], *W1n[2], *W2n[2];
    float *b1p[2], *qbias[2];
    for (int s = 0; s < 2; s++) {
        Wqn[s] = (u16*)alloc((size_t)DIM * DIM * 2);
        Wkn[s] = (u16*)alloc((size_t)DIM * DIM * 2);
        Wvn[s] = (u16*)alloc((size_t)DIM * DIM * 2);
        WoT[s] = (u16*)alloc((size_t)DIM * DIM * 2);
        W1n[s] = (u16*)alloc((size_t)FFP * DIM * 2);
        W2n[s] = (u16*)alloc((size_t)DIM * FFP * 2);
        b1p[s] = (float*)alloc((size_t)FFP * 4);
        qbias[s] = (float*)alloc((size_t)DIM * 4);
    }
    u16*   attW0b = (u16*)alloc((size_t)DIM * DIM * 2);
    float* pool0  = (float*)alloc((size_t)DIM * 4);
    float* pool1  = (float*)alloc((size_t)DIM * 4);

    auto gemm = [&](const u16* A, int lda, const u16* Bt, int ldb,
                    float* Cf, u16* Cb, int ldc, int M, int N, int K,
                    const float* bias, int relu) {
        dim3 g((N + BN - 1) / BN, M / BM, 1);
        gemm_bf16_wmma<<<g, 256, 0, stream>>>(A, lda, Bt, ldb, Cf, Cb, ldc, M, N, K, bias, relu);
    };

    // ---- weight prep (bf16, N x K row-major for the GEMM's transposed-B path)
    for (int s = 0; s < 2; s++) {
        prep_qkv_k<<<(DIM * DIM + 255) / 256, 256, 0, stream>>>(L[s].wq, L[s].wk, L[s].wv,
                                                                Wqn[s], Wkn[s], Wvn[s]);
        prep_qbias_k<<<DIM, 256, 0, stream>>>(L[s].wq, (s == 0) ? e1 : e2, qbias[s]);
        prep_wo_k<<<(DIM * DIM + 255) / 256, 256, 0, stream>>>(L[s].Wo, WoT[s]);
        prep_w1_k<<<(FFP * DIM + 255) / 256, 256, 0, stream>>>(L[s].W1, L[s].b1, W1n[s], b1p[s]);
        prep_w2_k<<<(DIM * FFP + 255) / 256, 256, 0, stream>>>(L[s].W2, W2n[s]);
    }
    // att_W0: Bt[d][k] = att_W0[d][k] -> plain convert
    f32_to_bf16_k<<<(DIM * DIM + 255) / 256, 256, 0, stream>>>(att_W0, attW0b, DIM * DIM);

    // ---- embeddings + positional encodings
    embed_pos_k<<<SEQ, 256, 0, stream>>>(contxtWords, e1p, word_embed, Xf[0]);
    embed_pos_k<<<SEQ, 256, 0, stream>>>(contxtWords, e2p, word_embed, Xf[1]);

    const float scale = 1.f / 16.f;   // 1/sqrt(HL)

    for (int s = 0; s < 2; s++) {
        float* X = Xf[s];
        f32_to_bf16_k<<<(SEQ * DIM + 255) / 256, 256, 0, stream>>>(X, Xb, SEQ * DIM);
        float* pooled = (s == 0) ? pool0 : pool1;

        for (int hop = 0; hop < 3; hop++) {
            // projections: Vq = X @ Wq^T + qbias (rank-1 from e), Vk, Vp
            gemm(Xb, DIM, Wqn[s], DIM, nullptr, Vq, DIM, SEQ, DIM, DIM, qbias[s], 0);
            gemm(Xb, DIM, Wkn[s], DIM, nullptr, Vk, DIM, SEQ, DIM, DIM, nullptr, 0);
            gemm(Xb, DIM, Wvn[s], DIM, nullptr, Vp, DIM, SEQ, DIM, DIM, nullptr, 0);
            // ctx GEMM needs Vp^T as its transposed-B operand
            transpose_bf16_k<<<dim3(DIM / 32, SEQ / 32), 256, 0, stream>>>(Vp, VpT, SEQ, DIM);

            for (int h = 0; h < NH; h++) {
                // scores[h] = Vq[:,hHL:] @ Vk[:,hHL:]^T ; Bt = Vk itself (N x K view)
                gemm(Vq + h * HLEN, DIM, Vk + h * HLEN, DIM,
                     scoresF, nullptr, SEQ, SEQ, SEQ, HLEN, nullptr, 0);
                softmax_row_k<<<SEQ, 256, 0, stream>>>(scoresF, probsB, scale);
                // ctx[h] = probs @ Vp[:,hHL:] ; Bt = VpT rows hHL.. (N=HL x K=S)
                gemm(probsB, SEQ, VpT + (size_t)h * HLEN * SEQ, SEQ,
                     nullptr, headsB + h * HLEN, DIM, SEQ, HLEN, SEQ, nullptr, 0);
            }
            // attn = heads @ Wo  (Bt = Wo^T)
            gemm(headsB, DIM, WoT[s], DIM, attnF, nullptr, DIM, SEQ, DIM, DIM, nullptr, 0);
            add_norm_k<<<SEQ, 256, 0, stream>>>(X, attnF, L[s].a2, L[s].b2, Yf, Yb);
            // FFN: H = relu(Y @ W1^T + b1) ; F = H @ W2^T + c2
            gemm(Yb, DIM, W1n[s], DIM, nullptr, Hb, FFP, SEQ, FFP, DIM, b1p[s], 1);
            gemm(Hb, FFP, W2n[s], FFP, ffnF, nullptr, DIM, SEQ, DIM, FFP, L[s].c2, 0);
            add_norm_k<<<SEQ, 256, 0, stream>>>(Yf, ffnF, L[s].a2, L[s].b2, X, Xb);
        }
        // pooling: A = X @ att_W0^T + att_b0; w = softmax_s(tanh(A)); pooled = sum_s X*w
        gemm(Xb, DIM, attW0b, DIM, attnF, nullptr, DIM, SEQ, DIM, DIM, att_b0, 0);
        pool_col_k<<<DIM, 256, 0, stream>>>(attnF, X, pooled);
    }

    final_fc_k<<<NCLS, 256, 0, stream>>>(sm_W, sm_b, pool0, pool1, relation, out);
}
